// LSTMModel_65541200937424
// MI455X (gfx1250) — matrix operand order repaired and checked by reading, not compiled, not run on hardware
//
#include <hip/hip_runtime.h>
#include <math.h>

typedef __attribute__((ext_vector_type(16))) __bf16 v16bf;
typedef __attribute__((ext_vector_type(8)))  __bf16 v8bf;
typedef __attribute__((ext_vector_type(8)))  float  v8f;

#define BB   512
#define TT   1024
#define FF   64
#define HH   128
#define NH   512    // 4*H
#define KTOT 192    // F + H
#define KP   200    // padded LDS row stride for A staging (bank-conflict-free)

__device__ __forceinline__ unsigned short f32_to_bf16(float f) {
    unsigned int u = __builtin_bit_cast(unsigned int, f);
    unsigned int r = u + 0x7FFFu + ((u >> 16) & 1u);
    return (unsigned short)(r >> 16);
}

__device__ __forceinline__ float tanh_fast(float x) {
#if __has_builtin(__builtin_amdgcn_tanhf)
    return __builtin_amdgcn_tanhf(x);
#elif __has_builtin(__builtin_amdgcn_tanh_f32)
    return __builtin_amdgcn_tanh_f32(x);
#else
    return tanhf(x);
#endif
}

// Persistent LSTM: one block = 16 batch rows for all T steps.
// 1024 threads = 32 wave32 waves; wave w owns gate-columns [16w, 16w+16).
// Weight B-fragments (time-invariant) live in VGPRs for the entire T loop.
// sched_group_barrier pipelines all 12 A-fragment ds_load_b128 ahead of the
// 6 chained v_wmma_f32_16x16x32_bf16 so DS latency is paid once per step.
__global__ __launch_bounds__(1024)
void lstm_persistent(const float* __restrict__ x,
                     const float* __restrict__ W,     // [FF][NH]
                     const float* __restrict__ U,     // [HH][NH]
                     const float* __restrict__ bias,  // [NH]
                     const float* __restrict__ Wfc,   // [HH]
                     const float* __restrict__ bfc,   // [1]
                     float* __restrict__ out)         // [BB]
{
    __shared__ unsigned short lA[16 * KP];   //  6400 B  (x_t || h_t) bf16
    __shared__ float zbuf[16 * NH];          // 32768 B  tanh(gate preacts)
    __shared__ float cbuf[16 * HH];          //  8192 B  cell state
    __shared__ float hbuf[16 * HH];          //  8192 B  hidden state (f32)

    const int tid  = threadIdx.x;
    const int wave = tid >> 5;
    const int lane = tid & 31;
    const int half = lane >> 4;
    const int l16  = lane & 15;
    const int b0   = blockIdx.x * 16;
    const int n0   = wave * 16;              // this wave's N-tile base

    // ---- one-time: load B fragments (weights) into registers ----
    // B layout (32x16 bf16): lane holds column N = n0+l16,
    // contiguous K = kc*32 + half*16 .. +15.
    v16bf wb[6];
    {
        const int n = n0 + l16;
        #pragma unroll
        for (int kc = 0; kc < 6; ++kc) {
            #pragma unroll
            for (int e = 0; e < 16; ++e) {
                int k = kc * 32 + half * 16 + e;
                float w = (k < FF) ? W[k * NH + n] : U[(k - FF) * NH + n];
                wb[kc][e] = __builtin_bit_cast(__bf16, f32_to_bf16(w));
            }
        }
    }
    const float bv = bias[n0 + l16];

    // ---- zero h region of A-staging and cell state ----
    for (int i = tid; i < 16 * HH; i += 1024) {
        cbuf[i] = 0.0f;
        int m = i >> 7, n = i & (HH - 1);
        lA[m * KP + FF + n] = (unsigned short)0;
    }
    // ---- stage x_0 (one float per thread: 16x64) ----
    {
        int m = tid >> 6, f = tid & 63;
        float v = x[((size_t)(b0 + m) * TT + 0) * FF + f];
        lA[m * KP + f] = f32_to_bf16(v);
    }

    const unsigned short* arow = lA + l16 * KP;

    for (int t = 0; t < TT; ++t) {
        __syncthreads();   // A-staging (x_t, h_t) ready

        // prefetch x_{t+1} into a register; consumed after next barrier
        const int tn = (t + 1 < TT) ? (t + 1) : t;
        const int xm = tid >> 6, xf = tid & 63;
        float xnext = x[((size_t)(b0 + xm) * TT + tn) * FF + xf];

        // ---- hoist ALL A fragments: 12 ds_load_b128 in flight together ----
        v16bf afrag[6];
        #pragma unroll
        for (int kc = 0; kc < 6; ++kc) {
            const int kbase = kc * 32;
            v8bf a_lo = *(const v8bf*)(arow + kbase + half * 16);
            v8bf a_hi = *(const v8bf*)(arow + kbase + half * 16 + 8);
            afrag[kc] = __builtin_shufflevector(a_lo, a_hi,
                          0,1,2,3,4,5,6,7,8,9,10,11,12,13,14,15);
        }

        // ---- z tile = [x_t|h_t] @ [W;U](:, n0:n0+16) + b ----
        v8f acc;
        #pragma unroll
        for (int r = 0; r < 8; ++r) acc[r] = bv;

        #pragma unroll
        for (int kc = 0; kc < 6; ++kc) {
            acc = __builtin_amdgcn_wmma_f32_16x16x32_bf16(
                      false, afrag[kc], false, wb[kc], (short)0, acc,
                      false, false);
        }

#if __has_builtin(__builtin_amdgcn_sched_group_barrier)
        // Enforce pipeline: all DS reads first, then the WMMA chain.
        // Masks: 0x100 = DS read, 0x008 = MFMA/WMMA.
        __builtin_amdgcn_sched_group_barrier(0x100, 12, 0);
        __builtin_amdgcn_sched_group_barrier(0x008,  6, 0);
#endif

        // ---- tanh(preacts) -> zbuf (C layout: M = r + 8*half, N = n0+l16)
        #pragma unroll
        for (int r = 0; r < 8; ++r) {
            int m = r + 8 * half;
            zbuf[m * NH + n0 + l16] = tanh_fast(acc[r]);
        }

        __syncthreads();   // all gates in zbuf

        // ---- gate combine: c,h update (2048 elems / 1024 threads) ----
        #pragma unroll
        for (int r = 0; r < 2; ++r) {
            int idx = tid + 1024 * r;        // = m*128 + n
            int m = idx >> 7, n = idx & (HH - 1);
            float iv = zbuf[m * NH + n];
            float fv = zbuf[m * NH + 128 + n];
            float gv = zbuf[m * NH + 256 + n];
            float ov = zbuf[m * NH + 384 + n];
            float c  = fv * cbuf[idx] + iv * gv;
            cbuf[idx] = c;
            float h  = ov * tanh_fast(c);
            hbuf[idx] = h;
            lA[m * KP + FF + n] = f32_to_bf16(h);   // stage h_{t+1}
        }
        // ---- stage x_{t+1} from the prefetched register ----
        if (t + 1 < TT) {
            lA[xm * KP + xf] = f32_to_bf16(xnext);
        }
    }

    __syncthreads();
    // ---- final projection: out[b] = h_last @ Wfc + bfc ----
    if (tid < 16) {
        float s = bfc[0];
        #pragma unroll 8
        for (int n = 0; n < HH; ++n) s += hbuf[tid * HH + n] * Wfc[n];
        out[b0 + tid] = s;
    }
}

extern "C" void kernel_launch(void* const* d_in, const int* in_sizes, int n_in,
                              void* d_out, int out_size, void* d_ws, size_t ws_size,
                              hipStream_t stream) {
    const float* x   = (const float*)d_in[0];
    const float* W   = (const float*)d_in[1];
    const float* U   = (const float*)d_in[2];
    const float* b   = (const float*)d_in[3];
    const float* Wfc = (const float*)d_in[4];
    const float* bfc = (const float*)d_in[5];
    float* out = (float*)d_out;
    (void)in_sizes; (void)n_in; (void)d_ws; (void)ws_size; (void)out_size;

    lstm_persistent<<<dim3(BB / 16), dim3(1024), 0, stream>>>(
        x, W, U, b, Wfc, bfc, out);
}
